// scaled_dot_31344671326317
// MI455X (gfx1250) — matrix operand order repaired and checked by reading, not compiled, not run on hardware
//
#include <hip/hip_runtime.h>
#include <stdint.h>

typedef __attribute__((ext_vector_type(16))) _Float16     v16h;
typedef __attribute__((ext_vector_type(8)))  _Float16     v8h;
typedef __attribute__((ext_vector_type(8)))  float        v8f;
typedef __attribute__((ext_vector_type(4)))  unsigned int u32x4;
typedef __attribute__((ext_vector_type(8)))  int          i32x8;
typedef __attribute__((ext_vector_type(4)))  int          i32x4;

#define SQ 4096
#define SK 4096
#define DD 1024
#define DVD 1024

#if __has_builtin(__builtin_amdgcn_tensor_load_to_lds) && __has_builtin(__builtin_amdgcn_s_wait_tensorcnt)
#define USE_TDM 1
#else
#define USE_TDM 0
#endif

// ---------------------------------------------------------------------------
// WMMA fragment loaders for v_wmma_f32_16x16x32_f16 (wave32).
// A (16x32 MxK) 16-bit, k order chosen here (B matches): lanes 0-15: row M=lane, K=0..15;
//                       lanes 16-31: row M=lane-16, K=16..31.
// B (32x16 KxN) 16-bit, same k order as A: lanes 0-15: col N=lane, K=0..15; lanes 16-31: K=16..31.
// C/D (16x16 f32): VGPR r -> M=r (lanes 0-15) / M=8+r (lanes 16-31), N=lane&15.
// ---------------------------------------------------------------------------
__device__ __forceinline__ v16h load_a_frag(const _Float16* base, int ld,
                                            int row0, int k0, int lane) {
    const int m  = lane & 15;
    const int hi = lane >> 4;
    const _Float16* p = base + (size_t)(row0 + m) * ld + (k0 + hi * 16);
    v8h lo = *(const v8h*)(p);
    v8h hv = *(const v8h*)(p + 8);
    v16h a;
#pragma unroll
    for (int i = 0; i < 8; ++i) { a[i] = lo[i]; a[i + 8] = hv[i]; }
    return a;
}

__device__ __forceinline__ v16h load_b_frag(const _Float16* base, int ld,
                                            int col0, int k0, int lane) {
    const int n  = lane & 15;
    const int hi = lane >> 4;
    return *(const v16h*)(base + (size_t)(col0 + n) * ld + (k0 + hi * 16));
}

#if USE_TDM
// ---------------------------------------------------------------------------
// Issue a TDM 2D tile load: Global (row-major, row pitch = row_len elems of
// 2 bytes) -> LDS (packed tile_rows x 32 f16). D# built per CDNA5 ISA 8.3/8.4.
// This toolchain's builtin takes 6 args (g0, g1, g2, g3, extra, cpol).
// ---------------------------------------------------------------------------
__device__ __forceinline__ void tdm_load_tile_2d(unsigned int lds_off,
                                                 const _Float16* gptr,
                                                 int tile_rows, int row_len,
                                                 int tensor_rows) {
    unsigned long long ga = (unsigned long long)(uintptr_t)gptr;
    u32x4 g0;
    g0[0] = 1u;                                        // count=1 (valid), user mode
    g0[1] = lds_off;                                   // lds_addr (bytes)
    g0[2] = (unsigned int)(ga & 0xffffffffull);        // global_addr[31:0]
    g0[3] = (unsigned int)((ga >> 32) & 0x01ffffffull) // global_addr[56:32]
            | (2u << 30);                              // type=2 ("image")
    i32x8 g1;
    g1[0] = 0x00010000;                                // wg_mask=0, data_size=1 (2B)
    g1[1] = (int)(((unsigned)row_len & 0xffffu) << 16);            // tensor_dim0[15:0]
    g1[2] = (int)((((unsigned)row_len >> 16) & 0xffffu)            // tensor_dim0[31:16]
            | (((unsigned)tensor_rows & 0xffffu) << 16));          // tensor_dim1[15:0]
    g1[3] = (int)((((unsigned)tensor_rows >> 16) & 0xffffu)        // tensor_dim1[31:16]
            | (32u << 16));                                        // tile_dim0 = 32
    g1[4] = tile_rows & 0xffff;                        // tile_dim1 (tile_dim2=0)
    g1[5] = row_len;                                   // tensor_dim0_stride[31:0]
    g1[6] = 0;                                         // stride hi / dim1_stride
    g1[7] = 0;
    i32x4 z4 = {0, 0, 0, 0};
    i32x8 z8 = {0, 0, 0, 0, 0, 0, 0, 0};
    __builtin_amdgcn_tensor_load_to_lds(g0, g1, z4, z4, z8, 0);
}
#endif

// ---------------------------------------------------------------------------
// One 32-deep MMA step from staged LDS tiles: wave computes 64x64 (4x4 WMMA).
// ---------------------------------------------------------------------------
__device__ __forceinline__ void mma_step(const _Float16* As, const _Float16* Bs,
                                         int wr, int wc, int lane, v8f (&acc)[4][4]) {
    v16h a[4];
#pragma unroll
    for (int mt = 0; mt < 4; ++mt)
        a[mt] = load_a_frag(As, 32, wr * 64 + mt * 16, 0, lane);
#pragma unroll
    for (int nt = 0; nt < 4; ++nt) {
        v16h b = load_b_frag(Bs, 32, wc * 64 + nt * 16, 0, lane);
#pragma unroll
        for (int mt = 0; mt < 4; ++mt)
            acc[mt][nt] = __builtin_amdgcn_wmma_f32_16x16x32_f16(
                false, a[mt], false, b, (short)0, acc[mt][nt], false, false);
    }
}

// ---------------------------------------------------------------------------
// Tiled GEMM: C[r,c] = scale * sum_k A[r,k]*B[c,k]   (both row-major, pitch ldk)
// Block = 8 waves = 256x128 tile (wave grid 4x2, 64x64 per wave).
// A/B 32-deep K-slabs staged into LDS by the Tensor Data Mover (double buffer).
// ---------------------------------------------------------------------------
template <bool OUT_F16>
__global__ __launch_bounds__(256) void gemm_tdm(const _Float16* __restrict__ A,
                                                const _Float16* __restrict__ B,
                                                void* __restrict__ Cout,
                                                int ldk, int ldc,
                                                int b_tensor_rows, float scale) {
    __shared__ __align__(128) _Float16 sA[2][256 * 32];   // 2 x 16 KB
    __shared__ __align__(128) _Float16 sB[2][128 * 32];   // 2 x  8 KB

    const int lane = threadIdx.x & 31;
    const int wave = threadIdx.x >> 5;
    const int wr = wave & 3;               // 4 wave rows x 64
    const int wc = wave >> 2;              // 2 wave cols x 64
    const int qblk = blockIdx.y * 256;
    const int cblk = blockIdx.x * 128;
    const int nsteps = ldk / 32;

    v8f acc[4][4] = {};

#if USE_TDM
    const unsigned int ldsA0 = (unsigned int)(uintptr_t)&sA[0][0];
    const unsigned int ldsB0 = (unsigned int)(uintptr_t)&sB[0][0];
    if (wave == 0) {
        tdm_load_tile_2d(ldsA0, A + (size_t)qblk * ldk, 256, ldk, SQ);
        tdm_load_tile_2d(ldsB0, B + (size_t)cblk * ldk, 128, ldk, b_tensor_rows);
    }
    for (int it = 0; it < nsteps; ++it) {
        const int buf = it & 1;
        __builtin_amdgcn_s_wait_tensorcnt(0);   // no-op for waves with TENSORcnt==0
        __syncthreads();                        // staged tiles visible to all waves
        if (wave == 0 && (it + 1) < nsteps) {   // prefetch next slab into other buffer
            const int d0 = (it + 1) * 32;
            tdm_load_tile_2d(ldsA0 + (unsigned)(buf ^ 1) * (256u * 32u * 2u),
                             A + (size_t)qblk * ldk + d0, 256, ldk, SQ);
            tdm_load_tile_2d(ldsB0 + (unsigned)(buf ^ 1) * (128u * 32u * 2u),
                             B + (size_t)cblk * ldk + d0, 128, ldk, b_tensor_rows);
        }
        mma_step(&sA[buf][0], &sB[buf][0], wr, wc, lane, acc);
        __syncthreads();                        // all reads done before buf is reused
    }
#else
    for (int it = 0; it < nsteps; ++it) {
        const int d0 = it * 32;
        __syncthreads();
        {   // cooperative staging fallback: one row per thread
            const int t = threadIdx.x;
            const v8h* gA = (const v8h*)(A + (size_t)(qblk + t) * ldk + d0);
            v8h* dA = (v8h*)&sA[0][t * 32];
#pragma unroll
            for (int j = 0; j < 4; ++j) dA[j] = gA[j];
            if (t < 128) {
                const v8h* gB = (const v8h*)(B + (size_t)(cblk + t) * ldk + d0);
                v8h* dB = (v8h*)&sB[0][t * 32];
#pragma unroll
                for (int j = 0; j < 4; ++j) dB[j] = gB[j];
            }
        }
        __syncthreads();
        mma_step(&sA[0][0], &sB[0][0], wr, wc, lane, acc);
    }
#endif

    const int n  = lane & 15;
    const int hi = lane >> 4;
#pragma unroll
    for (int mt = 0; mt < 4; ++mt)
#pragma unroll
        for (int nt = 0; nt < 4; ++nt)
#pragma unroll
            for (int r = 0; r < 8; ++r) {
                const size_t row = (size_t)(qblk + wr * 64 + mt * 16 + r + hi * 8);
                const int    col = cblk + wc * 64 + nt * 16 + n;
                if (OUT_F16)
                    ((_Float16*)Cout)[row * ldc + col] = (_Float16)(acc[mt][nt][r] * scale);
                else
                    ((float*)Cout)[row * ldc + col] = acc[mt][nt][r] * scale;
            }
}

// ---------------------------------------------------------------------------
// Elementwise f32 -> f16 conversion (4 elements / thread).
// ---------------------------------------------------------------------------
__global__ void cvt_f16_kernel(const float* __restrict__ src,
                               _Float16* __restrict__ dst, int n4) {
    int i = blockIdx.x * 256 + threadIdx.x;
    if (i < n4) {
        float4 v = ((const float4*)src)[i];
        dst[4 * i + 0] = (_Float16)v.x;
        dst[4 * i + 1] = (_Float16)v.y;
        dst[4 * i + 2] = (_Float16)v.z;
        dst[4 * i + 3] = (_Float16)v.w;
    }
}

// ---------------------------------------------------------------------------
// Column-wise online softmax stats over q (softmax dim=0): partial + combine.
// ---------------------------------------------------------------------------
__global__ void col_partial(const _Float16* __restrict__ S,
                            float* __restrict__ pm, float* __restrict__ pl) {
    const int k  = blockIdx.x * 256 + threadIdx.x;
    const int rb = blockIdx.y;
    const int q0 = rb * 256;
    float m = -3.0e38f, l = 0.0f;
    for (int q = q0; q < q0 + 256; ++q) {
        float s  = (float)S[(size_t)q * SK + k];
        float mn = fmaxf(m, s);
        l = l * __expf(m - mn) + __expf(s - mn);
        m = mn;
    }
    pm[(size_t)rb * SK + k] = m;
    pl[(size_t)rb * SK + k] = l;
}

__global__ void col_final(const float* __restrict__ pm, const float* __restrict__ pl,
                          float* __restrict__ mcol, float* __restrict__ lcol) {
    const int k = blockIdx.x * 256 + threadIdx.x;
    float m = -3.0e38f;
#pragma unroll
    for (int r = 0; r < 16; ++r) m = fmaxf(m, pm[(size_t)r * SK + k]);
    float l = 0.0f;
#pragma unroll
    for (int r = 0; r < 16; ++r)
        l += pl[(size_t)r * SK + k] * __expf(pm[(size_t)r * SK + k] - m);
    mcol[k] = m;
    lcol[k] = l;
}

// ---------------------------------------------------------------------------
// VhT[d,k] = f16( V[k,d] / lcol[k] )  -- 32x32 LDS tile transpose.
// ---------------------------------------------------------------------------
__global__ void vt_scale(const float* __restrict__ V, const float* __restrict__ lcol,
                         _Float16* __restrict__ VhT) {
    __shared__ _Float16 tile[32][33];
    const int d0 = blockIdx.x * 32;
    const int k0 = blockIdx.y * 32;
    const int tx = threadIdx.x;
    const int ty = threadIdx.y;
#pragma unroll
    for (int j = 0; j < 32; j += 8) {
        const int k = k0 + ty + j;
        const float rl = 1.0f / lcol[k];
        tile[ty + j][tx] = (_Float16)(V[(size_t)k * DVD + d0 + tx] * rl);
    }
    __syncthreads();
#pragma unroll
    for (int j = 0; j < 32; j += 8)
        VhT[(size_t)(d0 + ty + j) * SK + (k0 + tx)] = tile[tx][ty + j];
}

// ---------------------------------------------------------------------------
// In-place: S[q,k] <- f16( exp(S[q,k] - mcol[k]) ). 8 contiguous elems/thread.
// ---------------------------------------------------------------------------
__global__ void expify(_Float16* __restrict__ S, const float* __restrict__ mcol) {
    const size_t i = ((size_t)blockIdx.x * 256 + threadIdx.x) * 8;
    const int k = (int)(i & (SK - 1));
#pragma unroll
    for (int j = 0; j < 8; ++j) {
        float s = (float)S[i + j];
        S[i + j] = (_Float16)__expf(s - mcol[k + j]);
    }
}

// ---------------------------------------------------------------------------
// Host launcher.
// ---------------------------------------------------------------------------
extern "C" void kernel_launch(void* const* d_in, const int* in_sizes, int n_in,
                              void* d_out, int out_size, void* d_ws, size_t ws_size,
                              hipStream_t stream) {
    (void)in_sizes; (void)n_in; (void)out_size; (void)ws_size;
    const float* Q = (const float*)d_in[0];
    const float* K = (const float*)d_in[1];
    const float* V = (const float*)d_in[2];
    float* out = (float*)d_out;

    char* ws = (char*)d_ws;
    constexpr size_t OFF_QH = 0;            // 8 MB  f16 Q
    constexpr size_t OFF_KH = 8388608;      // 8 MB  f16 K
    constexpr size_t OFF_VT = 16777216;     // 8 MB  f16 (V/l)^T
    constexpr size_t OFF_S  = 25165824;     // 32 MB f16 scores -> exp in place
    constexpr size_t OFF_MC = 58720256;     // 16 KB col max
    constexpr size_t OFF_LC = 58736640;     // 16 KB col sumexp
    constexpr size_t OFF_PM = 58753024;     // 256 KB partial max
    constexpr size_t OFF_PL = 59015168;     // 256 KB partial sumexp

    _Float16* Qh  = (_Float16*)(ws + OFF_QH);
    _Float16* Kh  = (_Float16*)(ws + OFF_KH);
    _Float16* VhT = (_Float16*)(ws + OFF_VT);
    _Float16* S   = (_Float16*)(ws + OFF_S);
    float* mcol   = (float*)(ws + OFF_MC);
    float* lcol   = (float*)(ws + OFF_LC);
    float* pm     = (float*)(ws + OFF_PM);
    float* pl     = (float*)(ws + OFF_PL);

    // 1) f32 -> f16 for Q and K.
    cvt_f16_kernel<<<4096, 256, 0, stream>>>(Q, Qh, (SQ * DD) / 4);
    cvt_f16_kernel<<<4096, 256, 0, stream>>>(K, Kh, (SK * DD) / 4);

    // 2) S = (1/SK) * Qh @ Kh^T   (f16 out, pitch SK)
    gemm_tdm<true><<<dim3(SK / 128, SQ / 256), 256, 0, stream>>>(
        Qh, Kh, (void*)S, DD, SK, SK, 1.0f / (float)SK);

    // 3) Column softmax stats over q (dim=0).
    col_partial<<<dim3(SK / 256, 16), 256, 0, stream>>>(S, pm, pl);
    col_final<<<SK / 256, 256, 0, stream>>>(pm, pl, mcol, lcol);

    // 4) VhT[d,k] = f16(V[k,d] / l[k]).
    vt_scale<<<dim3(DVD / 32, SK / 32), dim3(32, 8), 0, stream>>>(V, lcol, VhT);

    // 5) S <- exp(S - m[k]) in place.
    expify<<<(SQ * SK) / (256 * 8), 256, 0, stream>>>(S, mcol);

    // 6) out = E @ (V/l)^T   (f32 out, pitch DVD)
    gemm_tdm<false><<<dim3(DVD / 128, SQ / 256), 256, 0, stream>>>(
        S, VhT, (void*)out, SK, DVD, DVD, 1.0f);
}
